// InfinyAttention_80281528697327
// MI455X (gfx1250) — compile-verified, matrix operand order checked by reading, not run on hardware
//
#include <hip/hip_runtime.h>

// ---------------------------------------------------------------------------
// InfinyAttention for MI455X (gfx1250, wave32, WMMA).
// B=4, L=4096, D=1024, H=16, SC=8 -> DEPTH=64, S=512.
// All large GEMMs run through v_wmma_f32_16x16x32_bf16 (f32 accumulate).
// ---------------------------------------------------------------------------

typedef __attribute__((ext_vector_type(16))) __bf16 v16bf;
typedef __attribute__((ext_vector_type(8)))  float  v8f;
typedef __attribute__((ext_vector_type(4)))  __bf16 bf16x4;
typedef __attribute__((ext_vector_type(8)))  __bf16 bf16x8;

#define WMMA_BF16(a, b, c) \
  __builtin_amdgcn_wmma_f32_16x16x32_bf16(false, (a), false, (b), (short)0, (c), false, false)

__device__ __forceinline__ float eluf(float x) { return x >= 0.f ? x : expm1f(x); }
__device__ __forceinline__ float sigf(float x) { return 1.f / (1.f + __expf(-x)); }

// ===========================================================================
// GEMM: C[M=16384, N=1024] = A[M,1024] * W[1024,N] + bias
// mode 0: row-major output. mode 1: scatter to split-heads (B,sc,depth,S,H).
// 256 threads (8 waves); 128x128 C tile; each wave owns a 32x64 region
// (2x4 of 16x16 WMMA tiles). bf16 operands double-buffered in LDS.
// ===========================================================================
#define GK 1024
#define GN 1024

__device__ __forceinline__
void gp_load(const float* __restrict__ A, const float* __restrict__ W,
             int m0, int n0, int k0, int t, float4* aR, float4* bR)
{
#pragma unroll
  for (int i = 0; i < 4; ++i) {             // batch-issue 8 b128 loads
    int idx4 = i * 256 + t;                 // 0..1023 (float4 units)
    int ar = idx4 >> 3, ac = (idx4 & 7) << 2;
    aR[i] = *(const float4*)&A[(size_t)(m0 + ar) * GK + k0 + ac];
    int kk = idx4 >> 5, nn = (idx4 & 31) << 2;
    bR[i] = *(const float4*)&W[(size_t)(k0 + kk) * GN + n0 + nn];
  }
}

__device__ __forceinline__
void gp_store(__bf16 (*As)[40], __bf16 (*Bs)[40], int t,
              const float4* aR, const float4* bR)
{
#pragma unroll
  for (int i = 0; i < 4; ++i) {
    int idx4 = i * 256 + t;
    int ar = idx4 >> 3, ac = (idx4 & 7) << 2;
    bf16x4 av = { (__bf16)aR[i].x, (__bf16)aR[i].y, (__bf16)aR[i].z, (__bf16)aR[i].w };
    *(bf16x4*)&As[ar][ac] = av;             // ds_store_b64
    int kk = idx4 >> 5, nn = (idx4 & 31) << 2;
    Bs[nn + 0][kk] = (__bf16)bR[i].x;       // (N,K) layout: scalar stores
    Bs[nn + 1][kk] = (__bf16)bR[i].y;
    Bs[nn + 2][kk] = (__bf16)bR[i].z;
    Bs[nn + 3][kk] = (__bf16)bR[i].w;
  }
}

__global__ __launch_bounds__(256)
void gemm_proj(const float* __restrict__ A, const float* __restrict__ W,
               const float* __restrict__ bias, float* __restrict__ out, int mode)
{
  __shared__ __bf16 As[2][128][40];   // double-buffered, padded rows
  __shared__ __bf16 Bs[2][128][40];

  const int bm = blockIdx.x >> 3;
  const int bn = blockIdx.x & 7;
  const int m0 = bm * 128, n0 = bn * 128;
  const int t = threadIdx.x;
  const int lane = t & 31, w = t >> 5;
  const int wr = w >> 1, wc = w & 1;
  const int r = lane & 15, hi = lane >> 4;

  v8f acc[2][4] = {};
  float4 aR[4], bR[4];

  gp_load(A, W, m0, n0, 0, t, aR, bR);
  gp_store(As[0], Bs[0], t, aR, bR);
  int p = 0;

  for (int k0 = 0; k0 < GK; k0 += 32) {
    __syncthreads();                         // tile p visible to all waves
    const bool next = (k0 + 32) < GK;
    if (next) gp_load(A, W, m0, n0, k0 + 32, t, aR, bR);  // overlap w/ compute

    v16bf af[2], bfv[4];
#pragma unroll
    for (int mt = 0; mt < 2; ++mt) {
      const __bf16* pp = &As[p][wr * 32 + mt * 16 + r][hi * 8];
#pragma unroll
      for (int j = 0; j < 8; ++j) { af[mt][j] = pp[j]; af[mt][8 + j] = pp[16 + j]; }
    }
#pragma unroll
    for (int nt = 0; nt < 4; ++nt) {
      const __bf16* pp = &Bs[p][wc * 64 + nt * 16 + r][hi * 8];
#pragma unroll
      for (int j = 0; j < 8; ++j) { bfv[nt][j] = pp[j]; bfv[nt][8 + j] = pp[16 + j]; }
    }
#pragma unroll
    for (int mt = 0; mt < 2; ++mt)
#pragma unroll
      for (int nt = 0; nt < 4; ++nt)
        acc[mt][nt] = WMMA_BF16(af[mt], bfv[nt], acc[mt][nt]);

    if (next) gp_store(As[p ^ 1], Bs[p ^ 1], t, aR, bR);  // other buffer: no race
    p ^= 1;
  }

#pragma unroll
  for (int mt = 0; mt < 2; ++mt)
#pragma unroll
    for (int nt = 0; nt < 4; ++nt)
#pragma unroll
      for (int j = 0; j < 8; ++j) {
        int m = m0 + wr * 32 + mt * 16 + hi * 8 + j;
        int n = n0 + wc * 64 + nt * 16 + r;
        float vv = acc[mt][nt][j] + bias[n];
        if (mode == 0) {
          out[(size_t)m * GN + n] = vv;
        } else {
          int b = m >> 12, l = m & 4095;
          int c = l >> 9, si = l & 511;
          int hh = n >> 6, dd = n & 63;
          out[((size_t)(((b * 8 + c) * 64 + dd) << 9) + si) * 16 + hh] = vv;
        }
      }
}

// ===========================================================================
// Carry init / per-chunk copy (Mem: 65536 f32, z: 524288 f32), float4-wide
// ===========================================================================
__global__ __launch_bounds__(256)
void init_carry(const float* __restrict__ Memory0, const float* __restrict__ z0,
                float* __restrict__ mem, float* __restrict__ z)
{
  int i = blockIdx.x * 256 + threadIdx.x;    // float4 index
  if (i < 16384)  *(float4*)&mem[4 * i] = *(const float4*)&Memory0[(4 * i) & 255];
  if (i < 131072) *(float4*)&z[4 * i]   = *(const float4*)&z0[(4 * i) & 15];
}

__global__ __launch_bounds__(256)
void carry_copy(const float* __restrict__ mi, float* __restrict__ mo,
                const float* __restrict__ zi, float* __restrict__ zo)
{
  int i = blockIdx.x * 256 + threadIdx.x;
  if (i < 16384)  *(float4*)&mo[4 * i] = *(const float4*)&mi[4 * i];
  if (i < 131072) *(float4*)&zo[4 * i] = *(const float4*)&zi[4 * i];
}

// ===========================================================================
// Per-chunk attention + linear-memory scan step. One (b, dd, 64-row tile)
// per 256-thread block; grid = 4*64*8 = 2048. Mask is identically zero.
// ===========================================================================
__global__ __launch_bounds__(256)
void attn_chunk(const float* __restrict__ qbuf, const float* __restrict__ kbuf,
                const float* __restrict__ vbuf, const float* __restrict__ beta,
                const float* __restrict__ memIn, float* __restrict__ memOut,
                const float* __restrict__ zIn, float* __restrict__ zOut,
                float* __restrict__ OT, int c)
{
  __shared__ __bf16 kT[512][40];          // key rows, K padded 16->32
  __shared__ __bf16 vT[16][520];          // v transposed (N=h, K=s)
  __shared__ __bf16 P [64][520];          // logits / probabilities
  __shared__ float qrows[64][16];
  __shared__ float krows[64][16];
  __shared__ float lskS[64][16];          // elu(k) cached
  __shared__ float viP [64][16];
  __shared__ float Alin[64][16];
  __shared__ float dotS[64][16];
  __shared__ float MemS[16][16];
  __shared__ float dotPart[4][16][16];
  __shared__ float redbuf[64][4];
  __shared__ float rowsum[64];

  const int t = threadIdx.x;
  const int lane = t & 31, w = t >> 5;
  const int r = lane & 15, hi = lane >> 4;
  int bid = blockIdx.x;
  const int rt = bid & 7;  bid >>= 3;
  const int dd = bid & 63; bid >>= 6;
  const int b  = bid;
  const int s0 = rt * 64;

  const size_t base = (((size_t)b * 8 + c) * 64 + dd) * 512 * 16;
  const float* qc = qbuf + base;
  const float* kc = kbuf + base;
  const float* vc = vbuf + base;

  // ---- stage K / V^T (batched b128 loads, then convert+store) ----
  {
    float4 kR[8], vR[8];
#pragma unroll
    for (int i = 0; i < 8; ++i) {
      int i4 = i * 256 + t;                // 2048 float4s = 512x16
      int s = i4 >> 2, h4 = (i4 & 3) << 2;
      kR[i] = *(const float4*)&kc[s * 16 + h4];
      vR[i] = *(const float4*)&vc[s * 16 + h4];
    }
#pragma unroll
    for (int i = 0; i < 8; ++i) {
      int i4 = i * 256 + t;
      int s = i4 >> 2, h4 = (i4 & 3) << 2;
      bf16x4 kv = { (__bf16)kR[i].x, (__bf16)kR[i].y, (__bf16)kR[i].z, (__bf16)kR[i].w };
      *(bf16x4*)&kT[s][h4] = kv;
      vT[h4 + 0][s] = (__bf16)vR[i].x;
      vT[h4 + 1][s] = (__bf16)vR[i].y;
      vT[h4 + 2][s] = (__bf16)vR[i].z;
      vT[h4 + 3][s] = (__bf16)vR[i].w;
    }
  }
  for (int i = t; i < 1024; i += 256) {    // zero-pad K half, b128 stores
    int s = i >> 1, h8 = (i & 1) << 3;
    bf16x8 z8 = {};
    *(bf16x8*)&kT[s][16 + h8] = z8;
  }
  {                                        // q/k rows of this tile, f32
    int sl = t >> 2, h4 = (t & 3) << 2;
    *(float4*)&qrows[sl][h4] = *(const float4*)&qc[(s0 + sl) * 16 + h4];
    *(float4*)&krows[sl][h4] = *(const float4*)&kc[(s0 + sl) * 16 + h4];
  }
  MemS[t >> 4][t & 15] = memIn[(size_t)(b * 64 + dd) * 256 + t];
  __syncthreads();

  // ---- phase 1: logits = q @ k^T * 0.25 (4 row tiles x 2 col halves) ----
  const int mt1 = w & 3, ch = w >> 2;
  v16bf qf;
  {
    const float* pp = &qrows[mt1 * 16 + r][hi * 8];
#pragma unroll
    for (int j = 0; j < 8; ++j) { qf[j] = (__bf16)pp[j]; qf[8 + j] = (__bf16)0.f; }
  }
  for (int nt = 0; nt < 16; ++nt) {
    int n0 = ch * 256 + nt * 16;
    v16bf bfv;
    const __bf16* pp = &kT[n0 + r][hi * 8];
#pragma unroll
    for (int j = 0; j < 8; ++j) { bfv[j] = pp[j]; bfv[8 + j] = pp[16 + j]; }
    v8f cacc = {};
    cacc = WMMA_BF16(qf, bfv, cacc);
#pragma unroll
    for (int j = 0; j < 8; ++j)
      P[mt1 * 16 + hi * 8 + j][n0 + r] = (__bf16)(cacc[j] * 0.25f);
  }
  __syncthreads();

  // ---- phase 2: softmax (4 threads/row, bf16x8 vector LDS traffic) ----
  {
    const int row = t >> 2, part = t & 3;
    bf16x8* prow = (bf16x8*)&P[row][part * 128];
    float mx = -3.0e38f;
    for (int jj = 0; jj < 16; ++jj) {
      bf16x8 v = prow[jj];
#pragma unroll
      for (int q = 0; q < 8; ++q) mx = fmaxf(mx, (float)v[q]);
    }
    redbuf[row][part] = mx;
    __syncthreads();
    float m = fmaxf(fmaxf(redbuf[row][0], redbuf[row][1]),
                    fmaxf(redbuf[row][2], redbuf[row][3]));
    float sm = 0.f;
    for (int jj = 0; jj < 16; ++jj) {
      bf16x8 v = prow[jj], e;
#pragma unroll
      for (int q = 0; q < 8; ++q) {
        float ev = __expf((float)v[q] - m);
        sm += ev;
        e[q] = (__bf16)ev;
      }
      prow[jj] = e;
    }
    __syncthreads();                       // all max reads done
    redbuf[row][part] = sm;
    __syncthreads();
    if (part == 0)
      rowsum[row] = redbuf[row][0] + redbuf[row][1] + redbuf[row][2] + redbuf[row][3];
  }
  __syncthreads();

  // ---- phase 3: dot = P @ v (split-K: 4 row tiles x 2 K halves) ----
  {
    const int mt3 = w & 3, kh = w >> 2;
    v8f oacc = {};
    for (int kk = 0; kk < 8; ++kk) {
      int kb = kh * 256 + kk * 32;
      v16bf pa, vb;
      const __bf16* pp = &P[mt3 * 16 + r][kb + hi * 8];
      const __bf16* pv = &vT[r][kb + hi * 8];
#pragma unroll
      for (int j = 0; j < 8; ++j) {
        pa[j] = pp[j]; pa[8 + j] = pp[16 + j];
        vb[j] = pv[j]; vb[8 + j] = pv[16 + j];
      }
      oacc = WMMA_BF16(pa, vb, oacc);
    }
    if (kh == 1)
#pragma unroll
      for (int j = 0; j < 8; ++j) dotPart[mt3][hi * 8 + j][r] = oacc[j];
    __syncthreads();
    if (kh == 0)
#pragma unroll
      for (int j = 0; j < 8; ++j) {
        int sl = mt3 * 16 + hi * 8 + j;
        dotS[sl][r] = (oacc[j] + dotPart[mt3][hi * 8 + j][r]) / rowsum[sl];
      }
  }
  __syncthreads();

  // ---- phase 4a: A, vi', z' (one row per 4 threads, 4 cols each) ----
  {
    const int sl = t >> 2, h4 = (t & 3) << 2;
    float lq[16], lk[16];
#pragma unroll
    for (int j = 0; j < 16; ++j) { lq[j] = eluf(qrows[sl][j]); lk[j] = eluf(krows[sl][j]); }
    if ((t & 3) == 0)
#pragma unroll
      for (int j = 0; j < 16; ++j) lskS[sl][j] = lk[j];
    float aq0 = 0, aq1 = 0, aq2 = 0, aq3 = 0;
    float ak0 = 0, ak1 = 0, ak2 = 0, ak3 = 0;
#pragma unroll
    for (int j = 0; j < 16; ++j) {
      float m0_ = MemS[j][h4 + 0], m1_ = MemS[j][h4 + 1];
      float m2_ = MemS[j][h4 + 2], m3_ = MemS[j][h4 + 3];
      aq0 += lq[j] * m0_; aq1 += lq[j] * m1_; aq2 += lq[j] * m2_; aq3 += lq[j] * m3_;
      ak0 += lk[j] * m0_; ak1 += lk[j] * m1_; ak2 += lk[j] * m2_; ak3 += lk[j] * m3_;
    }
    size_t zb = ((size_t)dd * 512 + (s0 + sl)) * 16 + h4;
    float4 zv = *(const float4*)&zIn[zb];
    float4 vv = *(const float4*)&vc[(s0 + sl) * 16 + h4];
    Alin[sl][h4 + 0] = aq0 / (lq[h4 + 0] * zv.x);
    Alin[sl][h4 + 1] = aq1 / (lq[h4 + 1] * zv.y);
    Alin[sl][h4 + 2] = aq2 / (lq[h4 + 2] * zv.z);
    Alin[sl][h4 + 3] = aq3 / (lq[h4 + 3] * zv.w);
    viP[sl][h4 + 0] = vv.x - ak0 / (lk[h4 + 0] * zv.x);
    viP[sl][h4 + 1] = vv.y - ak1 / (lk[h4 + 1] * zv.y);
    viP[sl][h4 + 2] = vv.z - ak2 / (lk[h4 + 2] * zv.z);
    viP[sl][h4 + 3] = vv.w - ak3 / (lk[h4 + 3] * zv.w);
    atomicAdd(&zOut[zb + 0], lk[h4 + 0]);
    atomicAdd(&zOut[zb + 1], lk[h4 + 1]);
    atomicAdd(&zOut[zb + 2], lk[h4 + 2]);
    atomicAdd(&zOut[zb + 3], lk[h4 + 3]);
  }
  __syncthreads();

  // ---- phase 4b: Mem' += lsk^T @ vi' ; final gated output ----
  {
    int j = t >> 4, h = t & 15;
    float s = 0.f;
    for (int sl = 0; sl < 64; ++sl)
      s += lskS[sl][j] * viP[sl][h];
    atomicAdd(&memOut[(size_t)(b * 64 + dd) * 256 + j * 16 + h], s);
  }
  {
    const int sl = t >> 2, h4 = (t & 3) << 2;
    const int si = s0 + sl;
    float4 bt = *(const float4*)&beta[(((size_t)b * 64 + dd) * 512 + si) * 16 + h4];
    float gg[4] = { sigf(bt.x), sigf(bt.y), sigf(bt.z), sigf(bt.w) };
#pragma unroll
    for (int q = 0; q < 4; ++q) {
      int h = h4 + q;
      float o = gg[q] * Alin[sl][h] + (1.f - gg[q]) * dotS[sl][h];
      int lin   = si * 16 + h;
      int d_idx = (c * 64 + dd) * 2 + (lin >> 12);
      int l_idx = lin & 4095;
      OT[(((size_t)b << 12) + l_idx) * 1024 + d_idx] = o;
    }
  }
}

// ===========================================================================
// Host launcher
// ===========================================================================
extern "C" void kernel_launch(void* const* d_in, const int* in_sizes, int n_in,
                              void* d_out, int out_size, void* d_ws, size_t ws_size,
                              hipStream_t stream)
{
  (void)in_sizes; (void)n_in; (void)out_size; (void)ws_size;
  const float* query   = (const float*)d_in[0];
  const float* key_    = (const float*)d_in[1];
  const float* value   = (const float*)d_in[2];
  // d_in[3] = mask: identically zero * 1e-9 in reference -> no-op, skipped
  const float* Wq = (const float*)d_in[4];  const float* bq = (const float*)d_in[5];
  const float* Wk = (const float*)d_in[6];  const float* bk = (const float*)d_in[7];
  const float* Wv = (const float*)d_in[8];  const float* bv = (const float*)d_in[9];
  const float* Wo = (const float*)d_in[10]; const float* bo = (const float*)d_in[11];
  const float* Memory0 = (const float*)d_in[12];
  const float* z0      = (const float*)d_in[13];
  const float* beta    = (const float*)d_in[14];
  // d_in[15]=seq_count(8), d_in[16]=delta_rule(1): compile-time constants here

  float* ws  = (float*)d_ws;
  float* qb  = ws;                         // (B,sc,depth,S,H)  16,777,216 f32
  float* kb  = qb + 16777216;
  float* vb  = kb + 16777216;
  float* OT  = vb + 16777216;              // (B,L,D) pre-output
  float* mem = OT + 16777216;              // 2 x 65,536   (ping-pong)
  float* zz  = mem + 2 * 65536;            // 2 x 524,288  (ping-pong)

  dim3 blk(256);
  gemm_proj<<<1024, blk, 0, stream>>>(query, Wq, bq, qb, 1);
  gemm_proj<<<1024, blk, 0, stream>>>(key_,  Wk, bk, kb, 1);
  gemm_proj<<<1024, blk, 0, stream>>>(value, Wv, bv, vb, 1);
  init_carry<<<512, blk, 0, stream>>>(Memory0, z0, mem, zz);

  for (int c = 0; c < 8; ++c) {            // sequential scan over chunks
    float* mi = mem + (c & 1) * 65536;
    float* mo = mem + ((c + 1) & 1) * 65536;
    float* zi = zz  + (c & 1) * 524288;
    float* zo = zz  + ((c + 1) & 1) * 524288;
    carry_copy<<<512, blk, 0, stream>>>(mi, mo, zi, zo);
    attn_chunk<<<2048, blk, 0, stream>>>(qb, kb, vb, beta, mi, mo, zi, zo, OT, c);
  }

  gemm_proj<<<1024, blk, 0, stream>>>(OT, Wo, bo, (float*)d_out, 0);
}
